// SynthesisLayer_39436389712356
// MI455X (gfx1250) — matrix operand order repaired and checked by reading, not compiled, not run on hardware
//
#include <hip/hip_runtime.h>
#include <hip/hip_bf16.h>

typedef __attribute__((ext_vector_type(16))) __bf16 v16bf;
typedef __attribute__((ext_vector_type(8)))  float  v8f;
typedef __attribute__((ext_vector_type(4)))  unsigned int u32x4;
typedef __attribute__((ext_vector_type(8)))  unsigned int u32x8;

#define NB    16
#define CIN   512
#define COUT  512
#define WD    512
#define HIN   32
#define RES   64
#define XUD   66
#define KTOT  4608          // Cin * 9 taps
#define INV_SQRT_WD 0.04419417382415922f
#define LRELU 0.2f
#define GAIN  1.4142135623730951f

// ---------------- styles: [16,512] = w @ A^T / sqrt(512) + bias ----------------
__global__ __launch_bounds__(256) void styles_kernel(
    const float* __restrict__ w, const float* __restrict__ aw,
    const float* __restrict__ ab, float* __restrict__ styles) {
  int gid = blockIdx.x * 256 + threadIdx.x;          // 8192 threads
  int b = gid >> 9, c = gid & 511;
  const float* wr = w + b * WD;
  const float* ar = aw + (size_t)c * WD;
  float acc = 0.f;
#pragma unroll 8
  for (int k = 0; k < WD; ++k) acc += wr[k] * ar[k];
  styles[gid] = acc * INV_SQRT_WD + ab[c];
}

// ------- modulate + demodulate + flip -> bf16 A-matrix [b][cout][tap*512+cin] -------
__global__ __launch_bounds__(256) void modulate_kernel(
    const float* __restrict__ weight, const float* __restrict__ styles,
    __bf16* __restrict__ wmod) {
  __shared__ float red[256];
  __shared__ float s_d;
  const int b  = blockIdx.x >> 9;
  const int co = blockIdx.x & 511;
  const float* wbase = weight + (size_t)co * KTOT;   // [cin][kh][kw] flat
  const float* st    = styles + b * CIN;

  float acc = 0.f;
  for (int i = threadIdx.x; i < KTOT; i += 256) {    // coalesced read
    float m = wbase[i] * st[i / 9];
    acc += m * m;
  }
  red[threadIdx.x] = acc;
  __syncthreads();
  for (int s = 128; s > 0; s >>= 1) {
    if (threadIdx.x < s) red[threadIdx.x] += red[threadIdx.x + s];
    __syncthreads();
  }
  if (threadIdx.x == 0) s_d = rsqrtf(red[0] + 1e-8f);
  __syncthreads();
  const float d = s_d;

  __bf16* out = wmod + (size_t)(b * COUT + co) * KTOT;
  for (int j = threadIdx.x; j < KTOT; j += 256) {    // coalesced write
    int tap_d = j >> 9;
    int cin   = j & 511;
    int src   = cin * 9 + (8 - tap_d);               // spatial flip ::-1,::-1
    out[j] = (__bf16)(wbase[src] * st[cin] * d);
  }
}

// ------- upfirdn2d up=2, pad=3, separable [1,3,3,1]/8 -> bf16 NHWC [b][66][66][cin] -------
__global__ __launch_bounds__(256) void upsample_kernel(
    const float* __restrict__ x, __bf16* __restrict__ xu) {
  const int xo = blockIdx.x, yo = blockIdx.y, b = blockIdx.z;
  const float f4[4] = {0.125f, 0.375f, 0.375f, 0.125f};
  int iy[2], ix[2]; float wy[2], wx[2];
  int ny = 0, nx = 0;
#pragma unroll
  for (int p = 0; p < 4; ++p) {
    int t = yo + p - 3;
    if (t >= 0 && (t & 1) == 0 && (t >> 1) < HIN) { iy[ny] = t >> 1; wy[ny] = f4[p]; ++ny; }
  }
#pragma unroll
  for (int q = 0; q < 4; ++q) {
    int t = xo + q - 3;
    if (t >= 0 && (t & 1) == 0 && (t >> 1) < HIN) { ix[nx] = t >> 1; wx[nx] = f4[q]; ++nx; }
  }
  __bf16* orow = xu + ((size_t)(b * XUD + yo) * XUD + xo) * CIN;
  for (int c = threadIdx.x; c < CIN; c += 256) {
    const float* xb = x + (size_t)(b * CIN + c) * (HIN * HIN);
    float acc = 0.f;
    for (int a = 0; a < ny; ++a)
      for (int e = 0; e < nx; ++e)
        acc += wy[a] * wx[e] * xb[iy[a] * HIN + ix[e]];
    orow[c] = (__bf16)(acc * 4.0f);                  // * UP^2
  }
}

// ---- TDM: issue one 2-D tile load (tile = tile1 rows x 32 bf16, row stride in elements).
// D# per ISA 8.3/8.4: group0 = {count=1, lds_addr, global_addr, type=2},
// group1 = {data_size=2B, tensor_dim0=32, tensor_dim1=tile1, tile_dim0=32,
//           tile_dim1=tile1, tensor_dim0_stride=stride}. Groups 2/3 = NULL (2-D).
__device__ __forceinline__ void tdm_load_2d(unsigned lds_off, const void* gptr,
                                            unsigned tile1, unsigned stride) {
  unsigned long long ga = (unsigned long long)gptr;
  u32x4 g0;
  g0[0] = 1u;                                        // count=1, user mode
  g0[1] = lds_off;                                   // lds_addr
  g0[2] = (unsigned)ga;                              // global_addr[31:0]
  g0[3] = (unsigned)((ga >> 32) & 0x01FFFFFFu) | (2u << 30);  // addr[56:32] | type=2
  u32x8 g1;
  g1[0] = 1u << 16;                                  // data_size = 1 -> 2 bytes
  g1[1] = 32u << 16;                                 // tensor_dim0 = 32 (lo16)
  g1[2] = (tile1 & 0xFFFFu) << 16;                   // tensor_dim0 hi=0 | tensor_dim1 lo16
  g1[3] = 32u << 16;                                 // tensor_dim1 hi=0 | tile_dim0 = 32
  g1[4] = tile1 & 0xFFFFu;                           // tile_dim1 | tile_dim2=0
  g1[5] = stride;                                    // tensor_dim0_stride lo32
  g1[6] = 0u;                                        // stride hi | dim1_stride lo
  g1[7] = 0u;
  asm volatile("tensor_load_to_lds %0, %1" :: "s"(g0), "s"(g1) : "memory");
}

// ---------------- main conv as implicit GEMM via v_wmma_f32_16x16x32_bf16 ----------------
// Per batch: M=512 (cout), N=4096 (64x64), K=4608. Block: 256 thr (8 waves).
// Tile M0=128, N0=128 (2 output rows), K chunk 32, double-buffered LDS fed by TDM.
// wave(wm 0..3, wn 0..1): 32 M x 64 N -> 2 A-frags x 4 B-frags = 8 WMMA / k-step.
__global__ __launch_bounds__(256) void conv_kernel(
    const __bf16* __restrict__ wmod,      // [b][cout][tap*512+cin]
    const __bf16* __restrict__ xu,        // [b][66][66][cin]  (NHWC bf16)
    const float* __restrict__ bias,
    const float* __restrict__ noise,
    const float* __restrict__ noise_strength,
    float* __restrict__ out) {
  __shared__ __align__(16) __bf16 ldsA[2][128 * 32];   // [m][k]  2 x 8 KB
  __shared__ __align__(16) __bf16 ldsB[2][128 * 32];   // [n][k]  2 x 8 KB

  const int tile_n = blockIdx.x;         // 0..31 -> 128 spatial
  const int m0     = blockIdx.y * 128;   // cout base
  const int b      = blockIdx.z;
  const int tid    = threadIdx.x;
  const int lane   = tid & 31;
  const int wave   = tid >> 5;
  const int wm     = wave >> 1;          // 0..3 : 32 couts each
  const int wn     = wave & 1;           // 0..1 : 64 cols each
  const int oh0    = tile_n * 2;

  const int khalf = lane >> 4;
  const int lanem = lane & 15;

  unsigned ldsA_off[2], ldsB_off[2];
#pragma unroll
  for (int i = 0; i < 2; ++i) {
    ldsA_off[i] = (unsigned)(unsigned long long)(const void*)&ldsA[i][0];
    ldsB_off[i] = (unsigned)(unsigned long long)(const void*)&ldsB[i][0];
  }

  auto issue = [&](int kcn, int nb) {
    const int tap = kcn >> 4, cin0 = (kcn & 15) * 32;
    const int kh = tap / 3, kw = tap % 3;
    // A: 128 cout-rows x 32 cin, row stride KTOT elements
    tdm_load_2d(ldsA_off[nb],
                wmod + ((size_t)(b * COUT + m0) * KTOT + tap * 512 + cin0),
                128u, (unsigned)KTOT);
    // B: two input rows, each 64 spatial points x 32 cin, point stride CIN elements
    tdm_load_2d(ldsB_off[nb],
                xu + ((size_t)((b * XUD + oh0 + kh) * XUD) + kw) * CIN + cin0,
                64u, (unsigned)CIN);
    tdm_load_2d(ldsB_off[nb] + 64u * 32u * 2u,
                xu + ((size_t)((b * XUD + oh0 + 1 + kh) * XUD) + kw) * CIN + cin0,
                64u, (unsigned)CIN);
  };

  v8f acc[8] = {};
  union Frag { u32x4 u[2]; v16bf v; };

  if (wave == 0) issue(0, 0);

#pragma unroll 1
  for (int kc = 0; kc < 144; ++kc) {
    const int cur = kc & 1;
    if (wave == 0) __builtin_amdgcn_s_wait_tensorcnt(0);
    __syncthreads();                       // buffer `cur` ready for everyone
    if (wave == 0 && kc + 1 < 144) issue(kc + 1, cur ^ 1);  // overlap with WMMA below

    // A fragments: rows wm*32 + ma*16 + lanem; K {khalf*8..+7, 16+khalf*8..+7}
    Frag a[2];
#pragma unroll
    for (int ma = 0; ma < 2; ++ma) {
      const __bf16* ap = &ldsA[cur][(wm * 32 + ma * 16 + lanem) * 32];
      a[ma].u[0] = *reinterpret_cast<const u32x4*>(ap + khalf * 8);
      a[ma].u[1] = *reinterpret_cast<const u32x4*>(ap + 16 + khalf * 8);
    }
#pragma unroll
    for (int f = 0; f < 4; ++f) {
      const int ncol = wn * 64 + f * 16 + lanem;
      Frag bf;
      const __bf16* bp = &ldsB[cur][ncol * 32 + khalf * 8];
      bf.u[0] = *reinterpret_cast<const u32x4*>(bp);
      bf.u[1] = *reinterpret_cast<const u32x4*>(bp + 16);
#pragma unroll
      for (int ma = 0; ma < 2; ++ma)
        acc[ma * 4 + f] = __builtin_amdgcn_wmma_f32_16x16x32_bf16(
            false, a[ma].v, false, bf.v, (short)0, acc[ma * 4 + f], false, false);
    }
    __syncthreads();                       // all reads of `cur` done before reuse
  }

  // ---- epilogue: + noise*strength, + bias, leaky_relu, * gain ----
  const float ns = noise_strength[0];
#pragma unroll
  for (int f = 0; f < 4; ++f) {
    const int ng = tile_n * 128 + wn * 64 + f * 16 + lanem;
    const int oh = ng >> 6, ow = ng & 63;
    const float nz = noise[oh * RES + ow] * ns;
#pragma unroll
    for (int ma = 0; ma < 2; ++ma) {
#pragma unroll
      for (int r = 0; r < 8; ++r) {
        const int cout = m0 + wm * 32 + ma * 16 + (lane >> 4) * 8 + r;
        float v = acc[ma * 4 + f][r] + nz + bias[cout];
        v = (v > 0.f) ? v : LRELU * v;
        out[((size_t)(b * COUT + cout) << 12) + ng] = v * GAIN;
      }
    }
  }
}

// ---------------- launch ----------------
extern "C" void kernel_launch(void* const* d_in, const int* in_sizes, int n_in,
                              void* d_out, int out_size, void* d_ws, size_t ws_size,
                              hipStream_t stream) {
  const float* x   = (const float*)d_in[0];
  const float* w   = (const float*)d_in[1];
  const float* aw  = (const float*)d_in[2];
  const float* ab  = (const float*)d_in[3];
  const float* wt  = (const float*)d_in[4];
  const float* bs  = (const float*)d_in[5];
  const float* nzc = (const float*)d_in[6];
  const float* nst = (const float*)d_in[7];
  float* out = (float*)d_out;

  char* ws = (char*)d_ws;
  float*  styles = (float*)ws;                                   // 32 KB
  __bf16* wmod   = (__bf16*)(ws + 32768);                        // 72 MB
  __bf16* xu     = (__bf16*)(ws + 32768 + (size_t)NB * COUT * KTOT * 2);  // 68 MB

  styles_kernel  <<<32, 256, 0, stream>>>(w, aw, ab, styles);
  modulate_kernel<<<NB * COUT, 256, 0, stream>>>(wt, styles, wmod);
  upsample_kernel<<<dim3(XUD, XUD, NB), 256, 0, stream>>>(x, xu);
  conv_kernel    <<<dim3(32, 4, NB), 256, 0, stream>>>(wmod, xu, bs, nzc, nst, out);
}
